// BlockMLP_MixerBlock_27504970564324
// MI455X (gfx1250) — matrix operand order repaired and checked by reading, not compiled, not run on hardware
//
#include <hip/hip_runtime.h>
#include <hip/hip_bf16.h>

typedef float  v8f   __attribute__((ext_vector_type(8)));
typedef __bf16 v16bf __attribute__((ext_vector_type(16)));
typedef __bf16 v8bf  __attribute__((ext_vector_type(8)));

#define BS_  16384
#define DIM_ 2048
#define NB_  32
#define BD_  64
#define HD_  128

// ---- helpers ---------------------------------------------------------------

__device__ __forceinline__ v8bf load8cvt(const float* p) {
  float4 a = *(const float4*)p;
  float4 b = *(const float4*)(p + 4);
  v8bf r;
  r[0] = (__bf16)a.x; r[1] = (__bf16)a.y; r[2] = (__bf16)a.z; r[3] = (__bf16)a.w;
  r[4] = (__bf16)b.x; r[5] = (__bf16)b.y; r[6] = (__bf16)b.z; r[7] = (__bf16)b.w;
  return r;
}
__device__ __forceinline__ v8bf load8cvt(const __bf16* p) { return *(const v8bf*)p; }

__device__ __forceinline__ v16bf cat16(v8bf lo, v8bf hi) {
  v16bf r;
#pragma unroll
  for (int i = 0; i < 8; ++i) { r[i] = lo[i]; r[i + 8] = hi[i]; }
  return r;
}

__device__ __forceinline__ void store_out(float* p, float v)  { *p = v; }
__device__ __forceinline__ void store_out(__bf16* p, float v) { *p = (__bf16)v; }

__device__ __forceinline__ float gelu_erf(float v) {
  return 0.5f * v * (1.0f + erff(v * 0.70710678118654752f));
}

// ---- fused block-MLP layer (one feature-block x 64-row tile per WG) --------
//
// WMMA operand layouts per CDNA5 ISA 7.12.2 (wave32):
//   A (16x32 bf16): lane m=L&15, hi=L>>4; elems 0..7 -> K=hi*8+e, elems 8..15 -> K=16+hi*8+e
//   B (32x16 bf16): lane n=L&15; same K order as A (elems 0..7 -> K=hi*8+e, 8..15 -> K=16+hi*8+e)
//   C/D (16x16 f32): elem j -> M=j+8*hi, N=L&15

template <typename TIN, typename TOUT>
__global__ __launch_bounds__(128) void block_mlp_layer(
    const TIN* __restrict__ x, const float* __restrict__ W1,
    const float* __restrict__ b1, const float* __restrict__ W2,
    const float* __restrict__ b2, TOUT* __restrict__ out) {
  __shared__ __bf16 w1t[HD_][72];      // w1t[o][i] = W1[n][i][o], padded row
  __shared__ __bf16 w2t[BD_][136];     // w2t[o][h] = W2[n][h][o], padded row
  __shared__ __bf16 hbuf[4][16][136];  // per-wave gelu(h) staging, padded row

  const int tid  = threadIdx.x;
  const int lane = tid & 31;
  const int wv   = tid >> 5;
  const int nblk = blockIdx.x;
  const int m0   = blockIdx.y * 64 + wv * 16;

  // Stage weights (f32 global -> bf16 LDS, transposed so K runs contiguously)
  const float* W1g = W1 + (size_t)nblk * (BD_ * HD_);
#pragma unroll 4
  for (int e = tid; e < BD_ * HD_; e += 128)
    w1t[e & (HD_ - 1)][e >> 7] = (__bf16)W1g[e];
  const float* W2g = W2 + (size_t)nblk * (HD_ * BD_);
#pragma unroll 4
  for (int e = tid; e < HD_ * BD_; e += 128)
    w2t[e & (BD_ - 1)][e >> 6] = (__bf16)W2g[e];
  __syncthreads();

  const int m  = lane & 15;
  const int hi = lane >> 4;

  const TIN* arow = x + (size_t)(m0 + m) * DIM_ + nblk * BD_;

  // ---- GEMM1: [16x64] @ [64x128] -> 8 N-tiles, K in 2 chunks of 32 ----
  v8f acc[8];
#pragma unroll
  for (int nt = 0; nt < 8; ++nt)
#pragma unroll
    for (int j = 0; j < 8; ++j) acc[nt][j] = 0.0f;

#pragma unroll
  for (int kc = 0; kc < 2; ++kc) {
    v16bf a = cat16(load8cvt(arow + kc * 32 + hi * 8),
                    load8cvt(arow + kc * 32 + 16 + hi * 8));
#pragma unroll
    for (int nt = 0; nt < 8; ++nt) {
      const __bf16* wp = &w1t[nt * 16 + m][kc * 32 + hi * 8];
      v16bf b = cat16(*(const v8bf*)wp, *(const v8bf*)(wp + 16));
      acc[nt] = __builtin_amdgcn_wmma_f32_16x16x32_bf16(
          false, a, false, b, (short)0, acc[nt], false, false);
    }
  }

  // ---- bias + exact-erf GELU, stage bf16 h into per-wave LDS tile ----
#pragma unroll
  for (int nt = 0; nt < 8; ++nt) {
    const int col   = nt * 16 + m;
    const float bia = b1[nblk * HD_ + col];
#pragma unroll
    for (int j = 0; j < 8; ++j)
      hbuf[wv][j + 8 * hi][col] = (__bf16)gelu_erf(acc[nt][j] + bia);
  }
  // same-wave DS ops are in-order; compiler inserts s_wait_dscnt for the RAW.

  // ---- GEMM2: [16x128] @ [128x64] -> 4 N-tiles, K in 4 chunks of 32 ----
  v8f oacc[4];
#pragma unroll
  for (int nt = 0; nt < 4; ++nt)
#pragma unroll
    for (int j = 0; j < 8; ++j) oacc[nt][j] = 0.0f;

#pragma unroll
  for (int kc = 0; kc < 4; ++kc) {
    const __bf16* hrow = &hbuf[wv][m][0];
    v16bf a = cat16(*(const v8bf*)(hrow + kc * 32 + hi * 8),
                    *(const v8bf*)(hrow + kc * 32 + 16 + hi * 8));
#pragma unroll
    for (int nt = 0; nt < 4; ++nt) {
      const __bf16* wp = &w2t[nt * 16 + m][kc * 32 + hi * 8];
      v16bf b = cat16(*(const v8bf*)wp, *(const v8bf*)(wp + 16));
      oacc[nt] = __builtin_amdgcn_wmma_f32_16x16x32_bf16(
          false, a, false, b, (short)0, oacc[nt], false, false);
    }
  }

  // ---- bias + store (lanes 0-15 / 16-31 each form 64B-contiguous runs) ----
#pragma unroll
  for (int nt = 0; nt < 4; ++nt) {
    const int col   = nblk * BD_ + nt * 16 + m;
    const float bia = b2[nblk * BD_ + nt * 16 + m];
#pragma unroll
    for (int j = 0; j < 8; ++j) {
      const int row = m0 + j + 8 * hi;
      store_out(out + (size_t)row * DIM_ + col, oacc[nt][j] + bia);
    }
  }
}

// ---- in-place butterfly permute: transpose each flat 4096-elem 64x64 tile --

__global__ __launch_bounds__(256) void transpose_tiles_f32(float* __restrict__ p) {
  __shared__ float t[64][65];
  const size_t base = (size_t)blockIdx.x * 4096;
  for (int i = threadIdx.x; i < 4096; i += 256) t[i >> 6][i & 63] = p[base + i];
  __syncthreads();
  for (int i = threadIdx.x; i < 4096; i += 256) p[base + i] = t[i & 63][i >> 6];
}

__global__ __launch_bounds__(256) void transpose_tiles_bf16(__bf16* __restrict__ p) {
  __shared__ __bf16 t[64][66];
  const size_t base = (size_t)blockIdx.x * 4096;
  for (int i = threadIdx.x; i < 4096; i += 256) t[i >> 6][i & 63] = p[base + i];
  __syncthreads();
  for (int i = threadIdx.x; i < 4096; i += 256) p[base + i] = t[i & 63][i >> 6];
}

// ---- launcher --------------------------------------------------------------

extern "C" void kernel_launch(void* const* d_in, const int* in_sizes, int n_in,
                              void* d_out, int out_size, void* d_ws, size_t ws_size,
                              hipStream_t stream) {
  const float* x    = (const float*)d_in[0];
  const float* W1_0 = (const float*)d_in[1];
  const float* b1_0 = (const float*)d_in[2];
  const float* W2_0 = (const float*)d_in[3];
  const float* b2_0 = (const float*)d_in[4];
  const float* W1_1 = (const float*)d_in[5];
  const float* b1_1 = (const float*)d_in[6];
  const float* W2_1 = (const float*)d_in[7];
  const float* b2_1 = (const float*)d_in[8];
  float*  out = (float*)d_out;
  __bf16* y0  = (__bf16*)d_ws;  // 16384*2048*2 = 64 MiB intermediate

  const dim3 grid(NB_, BS_ / 64);
  const int  ntiles = (BS_ * DIM_) / 4096;

  // Layer 0 (butterfly permutes are identity for i=0)
  block_mlp_layer<float, __bf16><<<grid, 128, 0, stream>>>(x, W1_0, b1_0, W2_0, b2_0, y0);
  // Layer-1 pre-permute: in-place 64x64 tile transpose (bf16)
  transpose_tiles_bf16<<<ntiles, 256, 0, stream>>>(y0);
  // Layer 1
  block_mlp_layer<__bf16, float><<<grid, 128, 0, stream>>>(y0, W1_1, b1_1, W2_1, b2_1, out);
  // Layer-1 post-permute: in-place 64x64 tile transpose (f32) on the output
  transpose_tiles_f32<<<ntiles, 256, 0, stream>>>(out);
}
